// DR_84164179132701
// MI455X (gfx1250) — matrix operand order repaired and checked by reading, not compiled, not run on hardware
//
#include <hip/hip_runtime.h>
#include <hip/hip_bf16.h>

#define NU  100000
#define NI  50000
#define NN  150000
#define EUI 1000000
#define EUU 500000
#define EII 500000
#define DD  64

typedef __attribute__((ext_vector_type(16))) _Float16 v16h;
typedef __attribute__((ext_vector_type(8)))  float    v8f;

// ---------------- helpers ----------------

__device__ inline void atomAddF(float* p, float v) {
  __hip_atomic_fetch_add(p, v, __ATOMIC_RELAXED, __HIP_MEMORY_SCOPE_AGENT);
}

__device__ inline size_t waveId() {
  return ((size_t)blockIdx.x * blockDim.x + threadIdx.x) >> 5;
}
__device__ inline size_t waveCount() {
  return ((size_t)gridDim.x * blockDim.x) >> 5;
}

__device__ inline void load8f(const float* __restrict__ p, float* d) {
  const float4 a = *reinterpret_cast<const float4*>(p);
  const float4 b = *reinterpret_cast<const float4*>(p + 4);
  d[0]=a.x; d[1]=a.y; d[2]=a.z; d[3]=a.w;
  d[4]=b.x; d[5]=b.y; d[6]=b.z; d[7]=b.w;
}

__device__ inline v8f wmma16(v16h a, v16h b, v8f c) {
  // D = A(16x32 f16) * B(32x16 f16) + C(16x16 f32)
  return __builtin_amdgcn_wmma_f32_16x16x32_f16(
      /*neg_a=*/false, a, /*neg_b=*/false, b,
      /*c_mod=*/(short)0, c, /*reuse_a=*/false, /*reuse_b=*/false);
}

// v_permlane16-based xor butterfly within each 16-lane row (VALU, stays off the LDS pipe)
__device__ inline float permx16(float v, unsigned s0, unsigned s1) {
  unsigned u = __builtin_bit_cast(unsigned, v);
  unsigned r = __builtin_amdgcn_permlane16(u, u, s0, s1, false, false);
  return __builtin_bit_cast(float, r);
}
__device__ inline float rsum16(float v) {
  v += permx16(v, 0x67452301u, 0xEFCDAB89u); // xor 1
  v += permx16(v, 0x54761032u, 0xDCFE98BAu); // xor 2
  v += permx16(v, 0x32107654u, 0xBA98FEDCu); // xor 4
  v += permx16(v, 0xFEDCBA98u, 0x76543210u); // xor 8
  return v;
}

// B tile: lane l holds column N = nt*16 + l%16; elements K = s*32 + 8*(l/16) + e and s*32 + 16 + 8*(l/16) + e, f32->f16
__device__ inline v16h load_b_tile(const float* __restrict__ M, int ld, int s, int nt) {
  const int lane = threadIdx.x & 31;
  const float* p = M + (size_t)(s * 32 + (lane >> 4) * 8) * ld + nt * 16 + (lane & 15);
  v16h b;
#pragma unroll
  for (int e = 0; e < 8; ++e) b[e] = (_Float16)p[(size_t)e * ld];
#pragma unroll
  for (int e = 0; e < 8; ++e) b[8 + e] = (_Float16)p[(size_t)(16 + e) * ld];
  return b;
}

// ---------------- elementwise / setup kernels ----------------

__global__ void k_init_x(const float* __restrict__ ue, const float* __restrict__ ie,
                         float* __restrict__ x, float* __restrict__ out) {
  const size_t n = (size_t)NN * DD;
  const size_t un = (size_t)NU * DD;
  for (size_t i = (size_t)blockIdx.x * blockDim.x + threadIdx.x; i < n;
       i += (size_t)gridDim.x * blockDim.x) {
    float v = (i < un) ? ue[i] : ie[i - un];
    x[i] = v;
    out[i] = v;   // acc starts at x
  }
}

__global__ void k_count_deg(const int* __restrict__ u, const int* __restrict__ it,
                            float* __restrict__ deg) {
  for (size_t e = (size_t)blockIdx.x * blockDim.x + threadIdx.x; e < (size_t)EUI;
       e += (size_t)gridDim.x * blockDim.x) {
    atomAddF(&deg[u[e]], 1.0f);
    atomAddF(&deg[NU + it[e]], 1.0f);
  }
}

__global__ void k_dinv(float* __restrict__ d, long n) {
  for (size_t i = (size_t)blockIdx.x * blockDim.x + threadIdx.x; i < (size_t)n;
       i += (size_t)gridDim.x * blockDim.x) {
    float v = d[i];
    d[i] = (v > 0.0f) ? rsqrtf(v) : 0.0f;
  }
}

__global__ void k_gval(const int* __restrict__ u, const int* __restrict__ it,
                       const float* __restrict__ dinv, float* __restrict__ gv) {
  for (size_t e = (size_t)blockIdx.x * blockDim.x + threadIdx.x; e < (size_t)EUI;
       e += (size_t)gridDim.x * blockDim.x)
    gv[e] = dinv[u[e]] * dinv[NU + it[e]];
}

__global__ void k_segsum(const int* __restrict__ h, const float* __restrict__ w,
                         float* __restrict__ s, long n) {
  for (size_t e = (size_t)blockIdx.x * blockDim.x + threadIdx.x; e < (size_t)n;
       e += (size_t)gridDim.x * blockDim.x)
    atomAddF(&s[h[e]], w[e]);
}

__global__ void k_recip(float* __restrict__ s, long n) {
  for (size_t i = (size_t)blockIdx.x * blockDim.x + threadIdx.x; i < (size_t)n;
       i += (size_t)gridDim.x * blockDim.x) {
    float v = s[i];
    s[i] = (v != 0.0f) ? (1.0f / v) : 0.0f;
  }
}

__global__ void k_scale(const int* __restrict__ h, const float* __restrict__ w,
                        const float* __restrict__ sinv, float* __restrict__ o, long n) {
  for (size_t e = (size_t)blockIdx.x * blockDim.x + threadIdx.x; e < (size_t)n;
       e += (size_t)gridDim.x * blockDim.x)
    o[e] = w[e] * sinv[h[e]];
}

__global__ void k_add(float* __restrict__ dst, const float* __restrict__ src, long n) {
  const size_t n4 = (size_t)n >> 2;
  float4* d4 = reinterpret_cast<float4*>(dst);
  const float4* s4 = reinterpret_cast<const float4*>(src);
  for (size_t i = (size_t)blockIdx.x * blockDim.x + threadIdx.x; i < n4;
       i += (size_t)gridDim.x * blockDim.x) {
    float4 a = d4[i]; const float4 b = s4[i];
    a.x += b.x; a.y += b.y; a.z += b.z; a.w += b.w;
    d4[i] = a;
  }
}

__global__ void k_l2norm(float* __restrict__ xb, long nrows) {
  const int lane = threadIdx.x & 31;
  for (size_t r = waveId(); r < (size_t)nrows; r += waveCount()) {
    float2* p = reinterpret_cast<float2*>(xb + r * DD) + lane;
    float2 v = *p;
    float s = rsum16(v.x * v.x + v.y * v.y);
    s += __shfl_xor(s, 16);
    const float inv = 1.0f / fmaxf(sqrtf(s), 1e-12f);
    v.x *= inv; v.y *= inv;
    *p = v;
  }
}

__global__ void k_combine(const float* __restrict__ Z, const float* __restrict__ b2,
                          const float* __restrict__ gs, const float* __restrict__ ts,
                          float* __restrict__ x, float* __restrict__ out) {
  const size_t n = (size_t)NN * DD;
  for (size_t i = (size_t)blockIdx.x * blockDim.x + threadIdx.x; i < n;
       i += (size_t)gridDim.x * blockDim.x) {
    const int col = (int)(i & 63);
    const float g = 1.0f / (1.0f + __expf(-(Z[i] + b2[col])));
    const float v = g * gs[i] + (1.0f - g) * ts[i];
    x[i] = v;
    out[i] += v;
  }
}

// ---------------- scatter SpMM kernels (wave per edge) ----------------

__global__ void k_spmm_g1(const int* __restrict__ u, const int* __restrict__ it,
                          const float* __restrict__ gv, const float* __restrict__ x,
                          float* __restrict__ out) {
  const int lane = threadIdx.x & 31;
  const size_t total = 2ull * EUI;
  for (size_t e = waveId(); e < total; e += waveCount()) {
    const size_t e0 = (e < (size_t)EUI) ? e : (e - EUI);
    const int uu = u[e0];
    const int ii = it[e0] + NU;
    const int h = (e < (size_t)EUI) ? uu : ii;
    const int t = (e < (size_t)EUI) ? ii : uu;
    const float v = gv[e0];
    const float2 xv = *reinterpret_cast<const float2*>(x + (size_t)t * DD + lane * 2);
    float* o = out + (size_t)h * DD + lane * 2;
    atomAddF(o, v * xv.x);
    atomAddF(o + 1, v * xv.y);
  }
}

__global__ void k_spmm_t1(const int* __restrict__ u, const int* __restrict__ it,
                          const float* __restrict__ wu, const float* __restrict__ wi,
                          const float* __restrict__ rinv, const float* __restrict__ x,
                          float* __restrict__ out) {
  const int lane = threadIdx.x & 31;
  const size_t total = 2ull * EUI;
  for (size_t e = waveId(); e < total; e += waveCount()) {
    const size_t e0 = (e < (size_t)EUI) ? e : (e - EUI);
    const int uu = u[e0];
    const int ii = it[e0] + NU;
    const int h = (e < (size_t)EUI) ? uu : ii;
    const int t = (e < (size_t)EUI) ? ii : uu;
    const float v = ((e < (size_t)EUI) ? wu[e0] : wi[e0]) * rinv[h];
    const float2 xv = *reinterpret_cast<const float2*>(x + (size_t)t * DD + lane * 2);
    float* o = out + (size_t)h * DD + lane * 2;
    atomAddF(o, v * xv.x);
    atomAddF(o + 1, v * xv.y);
  }
}

template <bool RINV>
__global__ void k_spmm_part(const int* __restrict__ h, const int* __restrict__ t,
                            const float* __restrict__ val, const float* __restrict__ rinv,
                            const float* __restrict__ x, float* __restrict__ out,
                            long E, int off) {
  const int lane = threadIdx.x & 31;
  for (size_t e = waveId(); e < (size_t)E; e += waveCount()) {
    const int hh = h[e];
    const int tt = t[e];
    float v = val[e];
    if (RINV) v *= rinv[hh];
    const float2 xv = *reinterpret_cast<const float2*>(x + (size_t)(tt + off) * DD + lane * 2);
    float* o = out + (size_t)(hh + off) * DD + lane * 2;
    atomAddF(o, v * xv.x);
    atomAddF(o + 1, v * xv.y);
  }
}

// ---------------- intent (WMMA) kernels ----------------
// 16 edges per wave: A = (h ⊙ t) 16x64 f16, B = intents 64x32 f16 (2 K-steps, 2 N-tiles)

struct DistOut {
  float d0[8], d1[8];   // softmax(dist): row r(+hi*8), cols q and 16+q
  float alpha8[8];      // alpha per row handled by this half
  int   uh8[8], th8[8]; // raw head / tail indices per row
};

template <bool DUAL>
__device__ inline void intent_dist16(long base, long E,
                                     const int* __restrict__ hIdx, const int* __restrict__ tIdx,
                                     int featHoff, int featToff,
                                     const float* __restrict__ x,
                                     const v16h (&B)[2][2], DistOut& o) {
  const int lane = threadIdx.x & 31;
  const int q = lane & 15, hi = lane >> 4;
  long be = base + q;
  if (be >= E) be = E - 1;
  const int hm = hIdx[be];
  const int tm = tIdx[be];
  const float* hp = x + (size_t)(hm + featHoff) * DD;
  const float* tp = x + (size_t)(tm + featToff) * DD;
  const int kbase = hi * 8;

  // Build A operands streaming: p = h*t, convert to f16 immediately, accumulate alpha sum.
  float part = 0.0f;
  v16h A0, A1;
#pragma unroll
  for (int c = 0; c < 4; ++c) {
    const int off = (c & 1) * 16 + (c >> 1) * 32 + kbase;
    float hv[8], tv[8];
    load8f(hp + off, hv);
    load8f(tp + off, tv);
#pragma unroll
    for (int e = 0; e < 8; ++e) {
      const float p = hv[e] * tv[e];
      part += p;
      if (c == 0)      A0[e]     = (_Float16)p;
      else if (c == 1) A0[8 + e] = (_Float16)p;
      else if (c == 2) A1[e]     = (_Float16)p;
      else             A1[8 + e] = (_Float16)p;
    }
  }
  // alpha = (sum(h*t) + 1)/2 ; lane covers 32 dims, pair with lane^16 for full 64
  float alpha = part + __shfl_xor(part, 16);
  alpha = (alpha + 1.0f) * 0.5f;

  v8f c0 = {0.f,0.f,0.f,0.f,0.f,0.f,0.f,0.f};
  v8f c1 = {0.f,0.f,0.f,0.f,0.f,0.f,0.f,0.f};
  c0 = wmma16(A0, B[0][0], c0); c0 = wmma16(A1, B[1][0], c0);
  c1 = wmma16(A0, B[0][1], c1); c1 = wmma16(A1, B[1][1], c1);

  // per-row softmax over 32 logits; logits are tiny (|.| << 80) so skip max-shift
  // (softmax is shift-invariant -> mathematically identical)
#pragma unroll
  for (int r = 0; r < 8; ++r) {
    const float e0 = __expf(c0[r]);
    const float e1 = __expf(c1[r]);
    const float s = rsum16(e0 + e1);
    const float inv = 1.0f / s;
    o.d0[r] = e0 * inv;
    o.d1[r] = e1 * inv;
  }
#pragma unroll
  for (int r = 0; r < 8; ++r) {
    const int src = r + hi * 8;   // edge row handled by this half
    o.alpha8[r] = __shfl(alpha, src);
    o.uh8[r] = __shfl(hm, src);
    if (DUAL) o.th8[r] = __shfl(tm, src);
  }
}

template <bool DUAL>
__global__ void k_intentA(const int* __restrict__ hIdx, const int* __restrict__ tIdx, long E,
                          int featHoff, int featToff,
                          const float* __restrict__ x, const float* __restrict__ intents,
                          float* __restrict__ agg, int aggHoff, int aggToff) {
  v16h B[2][2];
#pragma unroll
  for (int s = 0; s < 2; ++s)
#pragma unroll
    for (int nt = 0; nt < 2; ++nt) B[s][nt] = load_b_tile(intents, 32, s, nt);
  const int lane = threadIdx.x & 31;
  const int q = lane & 15, hi = lane >> 4;
  const size_t nb = ((size_t)E + 15) >> 4;
  for (size_t b = waveId(); b < nb; b += waveCount()) {
    const long base = (long)b * 16;
    DistOut o;
    intent_dist16<DUAL>(base, E, hIdx, tIdx, featHoff, featToff, x, B, o);
#pragma unroll
    for (int r = 0; r < 8; ++r) {
      const long m = base + r + hi * 8;
      if (m < E) {
        const float av = o.alpha8[r];
        float* a0 = agg + (size_t)(o.uh8[r] + aggHoff) * 32;
        atomAddF(a0 + q, av * o.d0[r]);
        atomAddF(a0 + 16 + q, av * o.d1[r]);
        if (DUAL) {
          float* a1 = agg + (size_t)(o.th8[r] + aggToff) * 32;
          atomAddF(a1 + q, av * o.d0[r]);
          atomAddF(a1 + 16 + q, av * o.d1[r]);
        }
      }
    }
  }
}

template <bool DUAL>
__global__ void k_intentB(const int* __restrict__ hIdx, const int* __restrict__ tIdx, long E,
                          int featHoff, int featToff,
                          const float* __restrict__ x, const float* __restrict__ intents,
                          const float* __restrict__ recip, int aggHoff, int aggToff,
                          float* __restrict__ wH, float* __restrict__ wT,
                          float* __restrict__ rowsum, int rsHoff, int rsToff) {
  v16h B[2][2];
#pragma unroll
  for (int s = 0; s < 2; ++s)
#pragma unroll
    for (int nt = 0; nt < 2; ++nt) B[s][nt] = load_b_tile(intents, 32, s, nt);
  const int lane = threadIdx.x & 31;
  const int q = lane & 15, hi = lane >> 4;
  const size_t nb = ((size_t)E + 15) >> 4;
  for (size_t b = waveId(); b < nb; b += waveCount()) {
    const long base = (long)b * 16;
    DistOut o;
    intent_dist16<DUAL>(base, E, hIdx, tIdx, featHoff, featToff, x, B, o);
#pragma unroll
    for (int r = 0; r < 8; ++r) {
      const float* rH = recip + (size_t)(o.uh8[r] + aggHoff) * 32;
      const float pu = rsum16(o.d0[r] * rH[q] + o.d1[r] * rH[16 + q]);
      const float wu = o.alpha8[r] * pu * (1.0f / 32.0f);
      float wi = 0.0f;
      if (DUAL) {
        const float* rT = recip + (size_t)(o.th8[r] + aggToff) * 32;
        const float pi = rsum16(o.d0[r] * rT[q] + o.d1[r] * rT[16 + q]);
        wi = o.alpha8[r] * pi * (1.0f / 32.0f);
      }
      const long m = base + r + hi * 8;
      if (q == 0 && m < E) {
        wH[m] = wu;
        atomAddF(&rowsum[o.uh8[r] + rsHoff], wu);
        if (DUAL) {
          wT[m] = wi;
          atomAddF(&rowsum[o.th8[r] + rsToff], wi);
        }
      }
    }
  }
}

// ---------------- gate GEMM: Z[150000,64] += f16(T[150000,64]) @ f16(W2sub[64,64]) ----------------

__global__ void k_gemm_acc(const float* __restrict__ T, const float* __restrict__ W2s,
                           float* __restrict__ Z) {
  const int wid = threadIdx.x >> 5;
  const int tile = blockIdx.x * (blockDim.x >> 5) + wid;
  const int nrowTiles = NN / 16;
  if (tile >= nrowTiles) return;
  const int lane = threadIdx.x & 31;
  const int q = lane & 15, hi = lane >> 4;

  const size_t m0 = (size_t)tile * 16;
  const float* arow = T + (m0 + q) * DD;
  const int kbase = hi * 8;

  // A operands (streamed conversion)
  v16h A0, A1;
  {
    float t[8];
    load8f(arow + kbase, t);
#pragma unroll
    for (int e = 0; e < 8; ++e) A0[e] = (_Float16)t[e];
    load8f(arow + kbase + 16, t);
#pragma unroll
    for (int e = 0; e < 8; ++e) A0[8 + e] = (_Float16)t[e];
    load8f(arow + 32 + kbase, t);
#pragma unroll
    for (int e = 0; e < 8; ++e) A1[e] = (_Float16)t[e];
    load8f(arow + 32 + kbase + 16, t);
#pragma unroll
    for (int e = 0; e < 8; ++e) A1[8 + e] = (_Float16)t[e];
  }

  // One N-tile at a time: keeps only 2 B tiles + 1 accumulator live.
#pragma unroll
  for (int nt = 0; nt < 4; ++nt) {
    const v16h B0 = load_b_tile(W2s, 64, 0, nt);
    const v16h B1 = load_b_tile(W2s, 64, 1, nt);
    v8f c;
#pragma unroll
    for (int r = 0; r < 8; ++r)
      c[r] = Z[(m0 + r + hi * 8) * DD + nt * 16 + q];
    c = wmma16(A0, B0, c);
    c = wmma16(A1, B1, c);
#pragma unroll
    for (int r = 0; r < 8; ++r)
      Z[(m0 + r + hi * 8) * DD + nt * 16 + q] = c[r];
  }
}

// ---------------- host orchestration ----------------

extern "C" void kernel_launch(void* const* d_in, const int* in_sizes, int n_in,
                              void* d_out, int out_size, void* d_ws, size_t ws_size,
                              hipStream_t stream) {
  (void)in_sizes; (void)n_in; (void)out_size; (void)ws_size;

  const float* user_emb = (const float*)d_in[0];
  const float* item_emb = (const float*)d_in[1];
  const float* intents  = (const float*)d_in[2];
  const float* W2       = (const float*)d_in[3];
  const float* b2       = (const float*)d_in[4];
  const float* uu_w     = (const float*)d_in[5];
  const float* ii_w     = (const float*)d_in[6];
  const int*   ui_u     = (const int*)d_in[7];
  const int*   ui_i     = (const int*)d_in[8];
  const int*   uu_h     = (const int*)d_in[9];
  const int*   uu_t     = (const int*)d_in[10];
  const int*   ii_h     = (const int*)d_in[11];
  const int*   ii_t     = (const int*)d_in[12];
  float* out = (float*)d_out;

  char* cur = (char*)d_ws;
  auto alloc = [&](size_t nElems) -> float* {
    float* p = (float*)cur;
    cur += ((nElems * sizeof(float)) + 255) & ~((size_t)255);
    return p;
  };
  const size_t ND = (size_t)NN * DD;
  float* x      = alloc(ND);
  float* Z      = alloc(ND);
  float* gsum   = alloc(ND);
  float* tsum   = alloc(ND);
  float* T      = alloc(ND);
  float* dinv   = alloc(NN);
  float* gval   = alloc(EUI);
  float* sU     = alloc(NU);
  float* sI     = alloc(NI);
  float* uuv    = alloc(EUU);
  float* iiv    = alloc(EII);
  float* agg    = alloc((size_t)NN * 32);
  float* rowsum = alloc(NN);
  float* wU     = alloc(EUI);
  float* wI     = alloc(EUI);

  const int TB  = 256;
  const int EWB = 4096;  // elementwise grid (grid-stride)
  const int SPB = 4096;  // spmm grid (wave per edge, grid-stride)
  const int INB = 2048;  // intent grid (wave per 16 edges, grid-stride)
  const int GEMM_BLOCKS = (NN / 16 + 3) / 4;

  // ---- init x and acc(out) ----
  k_init_x<<<EWB, TB, 0, stream>>>(user_emb, item_emb, x, out);

  // ---- degrees / symmetric-normalized G ----
  hipMemsetAsync(dinv, 0, NN * sizeof(float), stream);
  k_count_deg<<<SPB, TB, 0, stream>>>(ui_u, ui_i, dinv);
  k_dinv<<<EWB, TB, 0, stream>>>(dinv, NN);
  k_gval<<<EWB, TB, 0, stream>>>(ui_u, ui_i, dinv, gval);

  // ---- row-normalized uu_val / ii_val ----
  hipMemsetAsync(sU, 0, NU * sizeof(float), stream);
  k_segsum<<<EWB, TB, 0, stream>>>(uu_h, uu_w, sU, EUU);
  k_recip<<<EWB, TB, 0, stream>>>(sU, NU);
  k_scale<<<EWB, TB, 0, stream>>>(uu_h, uu_w, sU, uuv, EUU);
  hipMemsetAsync(sI, 0, NI * sizeof(float), stream);
  k_segsum<<<EWB, TB, 0, stream>>>(ii_h, ii_w, sI, EII);
  k_recip<<<EWB, TB, 0, stream>>>(sI, NI);
  k_scale<<<EWB, TB, 0, stream>>>(ii_h, ii_w, sI, iiv, EII);

  for (int layer = 0; layer < 2; ++layer) {
    hipMemsetAsync(Z, 0, ND * sizeof(float), stream);
    hipMemsetAsync(gsum, 0, ND * sizeof(float), stream);
    hipMemsetAsync(tsum, 0, ND * sizeof(float), stream);

    // ---- g1 = embed11(x) ----
    hipMemsetAsync(T, 0, ND * sizeof(float), stream);
    k_spmm_g1<<<SPB, TB, 0, stream>>>(ui_u, ui_i, gval, x, T);
    k_add<<<EWB, TB, 0, stream>>>(gsum, T, (long)ND);
    k_gemm_acc<<<GEMM_BLOCKS, 128, 0, stream>>>(T, W2 + 0 * 64 * 64, Z);

    // ---- g2 = embed12(x) ----
    hipMemsetAsync(T, 0, ND * sizeof(float), stream);
    k_spmm_part<false><<<SPB, TB, 0, stream>>>(uu_h, uu_t, uuv, nullptr, x, T, EUU, 0);
    k_spmm_part<false><<<SPB, TB, 0, stream>>>(ii_h, ii_t, iiv, nullptr, x, T, EII, NU);
    k_add<<<EWB, TB, 0, stream>>>(gsum, T, (long)ND);
    k_gemm_acc<<<GEMM_BLOCKS, 128, 0, stream>>>(T, W2 + 1 * 64 * 64, Z);

    // ---- t1 = embed21(x, x) ----
    hipMemsetAsync(agg, 0, (size_t)NN * 32 * sizeof(float), stream);
    k_intentA<true><<<INB, TB, 0, stream>>>(ui_u, ui_i, (long)EUI, 0, NU, x, intents, agg, 0, NU);
    k_recip<<<EWB, TB, 0, stream>>>(agg, (long)NN * 32);
    hipMemsetAsync(rowsum, 0, NN * sizeof(float), stream);
    k_intentB<true><<<INB, TB, 0, stream>>>(ui_u, ui_i, (long)EUI, 0, NU, x, intents,
                                            agg, 0, NU, wU, wI, rowsum, 0, NU);
    k_recip<<<EWB, TB, 0, stream>>>(rowsum, NN);
    hipMemsetAsync(T, 0, ND * sizeof(float), stream);
    k_spmm_t1<<<SPB, TB, 0, stream>>>(ui_u, ui_i, wU, wI, rowsum, x, T);
    k_l2norm<<<EWB, TB, 0, stream>>>(T, NN);
    k_add<<<EWB, TB, 0, stream>>>(tsum, T, (long)ND);
    k_gemm_acc<<<GEMM_BLOCKS, 128, 0, stream>>>(T, W2 + 2 * 64 * 64, Z);

    // ---- t2 = embed22(x, x) ----
    hipMemsetAsync(T, 0, ND * sizeof(float), stream);
    // uu part
    hipMemsetAsync(agg, 0, (size_t)NU * 32 * sizeof(float), stream);
    k_intentA<false><<<INB, TB, 0, stream>>>(uu_h, uu_t, (long)EUU, 0, 0, x, intents, agg, 0, 0);
    k_recip<<<EWB, TB, 0, stream>>>(agg, (long)NU * 32);
    hipMemsetAsync(rowsum, 0, NU * sizeof(float), stream);
    k_intentB<false><<<INB, TB, 0, stream>>>(uu_h, uu_t, (long)EUU, 0, 0, x, intents,
                                             agg, 0, 0, wU, nullptr, rowsum, 0, 0);
    k_recip<<<EWB, TB, 0, stream>>>(rowsum, NU);
    k_spmm_part<true><<<SPB, TB, 0, stream>>>(uu_h, uu_t, wU, rowsum, x, T, EUU, 0);
    // ii part
    hipMemsetAsync(agg, 0, (size_t)NI * 32 * sizeof(float), stream);
    k_intentA<false><<<INB, TB, 0, stream>>>(ii_h, ii_t, (long)EII, NU, NU, x, intents, agg, 0, 0);
    k_recip<<<EWB, TB, 0, stream>>>(agg, (long)NI * 32);
    hipMemsetAsync(rowsum, 0, NI * sizeof(float), stream);
    k_intentB<false><<<INB, TB, 0, stream>>>(ii_h, ii_t, (long)EII, NU, NU, x, intents,
                                             agg, 0, 0, wI, nullptr, rowsum, 0, 0);
    k_recip<<<EWB, TB, 0, stream>>>(rowsum, NI);
    k_spmm_part<true><<<SPB, TB, 0, stream>>>(ii_h, ii_t, wI, rowsum, x, T, EII, NU);
    k_l2norm<<<EWB, TB, 0, stream>>>(T, NN);
    k_add<<<EWB, TB, 0, stream>>>(tsum, T, (long)ND);
    k_gemm_acc<<<GEMM_BLOCKS, 128, 0, stream>>>(T, W2 + 3 * 64 * 64, Z);

    // ---- gate + combine + accumulate ----
    k_combine<<<EWB, TB, 0, stream>>>(Z, b2, gsum, tsum, x, out);
  }
}
